// CONNECT_86964497809993
// MI455X (gfx1250) — compile-verified, matrix operand order checked by reading, not run on hardware
//
#include <hip/hip_runtime.h>
#include <math.h>

#define N_NODES 100000
#define E_EVENTS 50000
#define C_COMM 256
#define M_DIM 128
#define D_DIM 128
#define F_DIM 128
#define T_DIM 64
#define MSG_DIM 448
#define G3 384
#define KCHUNK 4096
#define KT_IH 14           // 448 / 32
#define KT_ALL 18          // (448 + 128) / 32
#define NT_ALL 24          // 384 / 16
#define FRAG_ELEMS 512     // 32 lanes * 16 elems per (kt,nt) fragment block

typedef __attribute__((ext_vector_type(16))) __bf16 v16bf;
typedef __attribute__((ext_vector_type(16))) unsigned short v16us;
typedef __attribute__((ext_vector_type(8))) float v8f;

union FragU { v16us v; unsigned int u[8]; unsigned short s[16]; };

// 16-bit A-matrix fragment K index for element j (0..15), lane half hi (0/1).
__device__ __forceinline__ int frag_k(int j, int hi) {
  int v = j >> 1, h = j & 1;
  int base = (v < 4) ? (hi ? 8 : 0) : (hi ? 24 : 16);
  return base + ((v & 3) << 1) + h;
}

__device__ __forceinline__ unsigned short f2bf(float x) {
  unsigned int u = __float_as_uint(x);
  u += 0x7FFFu + ((u >> 16) & 1u); // round-to-nearest-even
  return (unsigned short)(u >> 16);
}

__device__ __forceinline__ float sigmoidf_(float x) { return 1.0f / (1.0f + __expf(-x)); }

// ---------------- init: zero argmax keys + community output ----------------
__global__ void k_init(unsigned long long* keys, float* out_comm) {
  int i = blockIdx.x * blockDim.x + threadIdx.x;
  int stride = gridDim.x * blockDim.x;
  for (int j = i; j < N_NODES; j += stride) keys[j] = 0ull;
  for (int j = i; j < C_COMM * M_DIM; j += stride) out_comm[j] = 0.0f;
}

// ---------------- 'last' aggregation via 64-bit atomicMax ----------------
__global__ void k_keys(const int* __restrict__ src, const int* __restrict__ dst,
                       const int* __restrict__ t, unsigned long long* __restrict__ keys) {
  int i = blockIdx.x * blockDim.x + threadIdx.x;
  if (i >= 2 * E_EVENTS) return;
  int e = (i < E_EVENTS) ? i : i - E_EVENTS;
  int node = (i < E_EVENTS) ? src[e] : dst[e];
  unsigned long long key =
      (((unsigned long long)(unsigned int)(t[e] + 1)) << 32) | (unsigned int)i;
  atomicMax(&keys[node], key);
}

// ---------- swizzle combined [W_ih ; W_hh] into bf16 B-fragment blocks ----------
// index: ((kt*24 + nt)*32 + lane)*16 + j
// value: kt<14 -> W_ih[kt*32 + frag_k(j, lane>>4)][nt*16 + lane&15]
//        kt>=14 -> W_hh[(kt-14)*32 + frag_k][...]
__global__ void k_prep(const float* __restrict__ W_ih, const float* __restrict__ W_hh,
                       unsigned short* __restrict__ wc_sw) {
  const int NC = KT_ALL * NT_ALL * FRAG_ELEMS; // 221184
  int idx = blockIdx.x * blockDim.x + threadIdx.x;
  if (idx >= NC) return;
  int j = idx & 15; int q = idx >> 4;
  int lane = q & 31; q >>= 5;
  int nt = q % NT_ALL, kt = q / NT_ALL;
  int kk = frag_k(j, lane >> 4);
  int n = nt * 16 + (lane & 15);
  float v;
  if (kt < KT_IH) v = W_ih[(kt * 32 + kk) * G3 + n];
  else            v = W_hh[((kt - KT_IH) * 32 + kk) * G3 + n];
  wc_sw[idx] = f2bf(v);
}

// ---------------- fused message-build + GRU via bf16 WMMA ----------------
// 32 nodes per workgroup, 256 threads = 8 waves.
// wave -> (row-tile rt = wv&1, column-group cg = wv>>1); cg owns matched
// r/z/n column tiles {2cg,2cg+1, 8+2cg,8+2cg+1, 16+2cg,16+2cg+1}.
__global__ void __launch_bounds__(256)
k_gru(const unsigned long long* __restrict__ keys,
      const int* __restrict__ t, const int* __restrict__ last_update,
      const float* __restrict__ event_feat, const float* __restrict__ src_emb,
      const float* __restrict__ dst_emb, const float* __restrict__ nodes_mem,
      const float* __restrict__ w_time, const float* __restrict__ b_time,
      const float* __restrict__ b_ih, const float* __restrict__ b_hh,
      const unsigned short* __restrict__ wc_sw,
      float* __restrict__ out_mem) {
  __shared__ unsigned short s_msg[32 * MSG_DIM];   // bf16 message tiles
  __shared__ unsigned short s_memb[32 * M_DIM];    // bf16 old memory
  __shared__ float s_memf[32 * M_DIM];             // f32 old memory (for blend)
  __shared__ int s_has[32];
  __shared__ int s_ev[32];
  __shared__ int s_half[32];
  __shared__ float s_dt[32];

  int tid = threadIdx.x;
  int node0 = blockIdx.x * 32;

  if (tid < 32) {
    unsigned long long key = keys[node0 + tid];
    int has = (key != 0ull);
    unsigned int idx = (unsigned int)(key & 0xffffffffu);
    int half = (has && idx >= (unsigned)E_EVENTS) ? 1 : 0;
    int e = has ? (half ? (int)idx - E_EVENTS : (int)idx) : 0;
    s_has[tid] = has;
    s_ev[tid] = e;
    s_half[tid] = half;
    s_dt[tid] = (float)(t[e] - last_update[node0 + tid]);
  }
  __syncthreads();

  // build message tiles: 32 rows x 448 cols
  for (int it = 0; it < (32 * MSG_DIM) / 256; ++it) {
    int idx = tid + it * 256;
    int row = idx / MSG_DIM;
    int col = idx - row * MSG_DIM;
    int e = s_ev[row], half = s_half[row];
    float val;
    if (col < 128) {
      const float* p = half ? dst_emb : src_emb;
      val = p[e * D_DIM + col];
    } else if (col < 256) {
      const float* p = half ? src_emb : dst_emb;
      val = p[e * D_DIM + (col - 128)];
    } else if (col < 384) {
      val = event_feat[e * F_DIM + (col - 256)];
    } else {
      int j = col - 384;
      val = cosf(s_dt[row] * w_time[j] + b_time[j]);
    }
    s_msg[idx] = f2bf(val);
  }
  // old memory tiles
  for (int it = 0; it < (32 * M_DIM) / 256; ++it) {
    int idx = tid + it * 256;
    int row = idx >> 7, col = idx & 127;
    float v = nodes_mem[(long)(node0 + row) * M_DIM + col];
    s_memf[idx] = v;
    s_memb[idx] = f2bf(v);
  }
  __syncthreads();

  int lane = tid & 31;
  int wv = tid >> 5;
  int rt = wv & 1;        // row tile (nodes rt*16 .. rt*16+15)
  int cg = wv >> 1;       // column group 0..3
  int m = rt * 16 + (lane & 15);
  int hi = lane >> 4;

  // accumulators: slots 0..3 = r0,r1,z0,z1 (shared between gx and gh),
  //               4..5 = xn0,xn1 (msg pass), 6..7 = hn0,hn1 (mem pass)
  v8f acc[8] = {};
  int ntab[6];
#pragma unroll
  for (int s = 0; s < 6; ++s) ntab[s] = (s >> 1) * 8 + 2 * cg + (s & 1);

  // per-slot weight fragment base pointers (constant displacement per kt)
  const unsigned short* wb[6];
#pragma unroll
  for (int s = 0; s < 6; ++s)
    wb[s] = wc_sw + (size_t)ntab[s] * FRAG_ELEMS + (size_t)lane * 16;

  // K-loop over message (kt 0..13)
#pragma unroll
  for (int kt = 0; kt < KT_IH; ++kt) {
    FragU a;
#pragma unroll
    for (int tp = 0; tp < 8; ++tp) {
      int k = kt * 32 + frag_k(2 * tp, hi);
      a.u[tp] = *(const unsigned int*)&s_msg[m * MSG_DIM + k];
    }
    v16bf av = __builtin_bit_cast(v16bf, a.v);
#pragma unroll
    for (int s = 0; s < 6; ++s) {
      v16us bu = *(const v16us*)(wb[s] + kt * (NT_ALL * FRAG_ELEMS));
      v16bf bv = __builtin_bit_cast(v16bf, bu);
      acc[s] = __builtin_amdgcn_wmma_f32_16x16x32_bf16(
          false, av, false, bv, (short)0, acc[s], false, false);
    }
  }
  // K-loop over old memory (kt 14..17): r/z accumulate into same slots, n separate
#pragma unroll
  for (int kt = KT_IH; kt < KT_ALL; ++kt) {
    FragU a;
#pragma unroll
    for (int tp = 0; tp < 8; ++tp) {
      int k = (kt - KT_IH) * 32 + frag_k(2 * tp, hi);
      a.u[tp] = *(const unsigned int*)&s_memb[m * M_DIM + k];
    }
    v16bf av = __builtin_bit_cast(v16bf, a.v);
#pragma unroll
    for (int s = 0; s < 6; ++s) {
      int slot = (s < 4) ? s : s + 2; // n-gate goes to slots 6,7
      v16us bu = *(const v16us*)(wb[s] + kt * (NT_ALL * FRAG_ELEMS));
      v16bf bv = __builtin_bit_cast(v16bf, bu);
      acc[slot] = __builtin_amdgcn_wmma_f32_16x16x32_bf16(
          false, av, false, bv, (short)0, acc[slot], false, false);
    }
  }

  // GRU gates fully in-register
#pragma unroll
  for (int s = 0; s < 2; ++s) {
    int colr = (2 * cg + s) * 16 + (lane & 15); // 0..127
    float br = b_ih[colr] + b_hh[colr];
    float bz = b_ih[colr + 128] + b_hh[colr + 128];
    float bni = b_ih[colr + 256];
    float bnh = b_hh[colr + 256];
#pragma unroll
    for (int p = 0; p < 8; ++p) {
      int mr = rt * 16 + p + (hi << 3);
      float r = sigmoidf_(acc[s][p] + br);
      float z = sigmoidf_(acc[2 + s][p] + bz);
      float nn = tanhf(acc[4 + s][p] + bni + r * (acc[6 + s][p] + bnh));
      float old = s_memf[mr * M_DIM + colr];
      float h = (1.0f - z) * nn + z * old;
      out_mem[(long)(node0 + mr) * M_DIM + colr] = s_has[mr] ? h : old;
    }
  }
}

// ---------------- comm_mem = incidence^T @ new_mem via bf16 WMMA ----------------
// new_mem (51 MB) stays L2-resident across the 16 c-tile passes (192 MB L2).
__global__ void __launch_bounds__(128)
k_comm(const float* __restrict__ incidence, const float* __restrict__ new_mem,
       float* __restrict__ out_comm) {
  __shared__ unsigned short s_incT[16 * 32];  // [c][k]
  __shared__ unsigned short s_memT[128 * 32]; // [m][k]
  int tid = threadIdx.x;
  int ct = blockIdx.x;     // c tile (16)
  int chunk = blockIdx.y;  // N chunk
  int lane = tid & 31;
  int wv = tid >> 5;
  int hi = lane >> 4;

  v8f acc0 = {}, acc1 = {};
  int mt0 = 2 * wv, mt1 = 2 * wv + 1;

  for (int step = 0; step < KCHUNK / 32; ++step) {
    int n0 = chunk * KCHUNK + step * 32;
    __syncthreads();
    for (int it = 0; it < 4; ++it) { // 32x16 incidence slab (transposed store)
      int idx = tid + it * 128;
      int nl = idx >> 4, c = idx & 15;
      int n = n0 + nl;
      float v = (n < N_NODES) ? incidence[(long)n * C_COMM + ct * 16 + c] : 0.0f;
      s_incT[c * 32 + nl] = f2bf(v);
    }
    for (int it = 0; it < 32; ++it) { // 32x128 memory slab (transposed store)
      int idx = tid + it * 128;
      int nl = idx >> 7, mc = idx & 127;
      int n = n0 + nl;
      float v = (n < N_NODES) ? new_mem[(long)n * M_DIM + mc] : 0.0f;
      s_memT[mc * 32 + nl] = f2bf(v);
    }
    __syncthreads();

    FragU a;
    int c = lane & 15;
#pragma unroll
    for (int tp = 0; tp < 8; ++tp) {
      int k = frag_k(2 * tp, hi);
      a.u[tp] = *(const unsigned int*)&s_incT[c * 32 + k];
    }
    v16bf av = __builtin_bit_cast(v16bf, a.v);
    FragU b0, b1;
#pragma unroll
    for (int tp = 0; tp < 8; ++tp) {
      int k = frag_k(2 * tp, hi);
      b0.u[tp] = *(const unsigned int*)&s_memT[(mt0 * 16 + (lane & 15)) * 32 + k];
      b1.u[tp] = *(const unsigned int*)&s_memT[(mt1 * 16 + (lane & 15)) * 32 + k];
    }
    acc0 = __builtin_amdgcn_wmma_f32_16x16x32_bf16(
        false, av, false, __builtin_bit_cast(v16bf, b0.v), (short)0, acc0, false, false);
    acc1 = __builtin_amdgcn_wmma_f32_16x16x32_bf16(
        false, av, false, __builtin_bit_cast(v16bf, b1.v), (short)0, acc1, false, false);
  }

#pragma unroll
  for (int p = 0; p < 8; ++p) {
    int cl = p + (hi << 3);
    atomicAdd(&out_comm[(ct * 16 + cl) * M_DIM + mt0 * 16 + (lane & 15)], acc0[p]);
    atomicAdd(&out_comm[(ct * 16 + cl) * M_DIM + mt1 * 16 + (lane & 15)], acc1[p]);
  }
}

extern "C" void kernel_launch(void* const* d_in, const int* in_sizes, int n_in,
                              void* d_out, int out_size, void* d_ws, size_t ws_size,
                              hipStream_t stream) {
  (void)in_sizes; (void)n_in; (void)out_size; (void)ws_size;
  const int* src = (const int*)d_in[0];
  const int* dst = (const int*)d_in[1];
  const int* t = (const int*)d_in[2];
  const int* last_update = (const int*)d_in[3];
  const float* event_feat = (const float*)d_in[4];
  const float* src_emb = (const float*)d_in[5];
  const float* dst_emb = (const float*)d_in[6];
  const float* nodes_mem = (const float*)d_in[7];
  const float* incidence = (const float*)d_in[8];
  const float* w_time = (const float*)d_in[9];
  const float* b_time = (const float*)d_in[10];
  const float* W_ih = (const float*)d_in[11];
  const float* W_hh = (const float*)d_in[12];
  const float* b_ih = (const float*)d_in[13];
  const float* b_hh = (const float*)d_in[14];

  float* out_mem = (float*)d_out;                          // [N, M]
  float* out_comm = out_mem + (size_t)N_NODES * M_DIM;     // [C, M]

  char* ws = (char*)d_ws;
  unsigned long long* keys = (unsigned long long*)ws;              // 800 KB
  unsigned short* wc_sw = (unsigned short*)(ws + (1 << 20));       // 442368 B

  k_init<<<256, 256, 0, stream>>>(keys, out_comm);
  k_keys<<<(2 * E_EVENTS + 255) / 256, 256, 0, stream>>>(src, dst, t, keys);
  k_prep<<<(KT_ALL * NT_ALL * FRAG_ELEMS + 255) / 256, 256, 0, stream>>>(W_ih, W_hh, wc_sw);
  k_gru<<<N_NODES / 32, 256, 0, stream>>>(keys, t, last_update, event_feat, src_emb,
                                          dst_emb, nodes_mem, w_time, b_time, b_ih, b_hh,
                                          wc_sw, out_mem);
  k_comm<<<dim3(16, (N_NODES + KCHUNK - 1) / KCHUNK), 128, 0, stream>>>(
      incidence, out_mem, out_comm);
}
